// Qwen3SimpleMoE_12429635355164
// MI455X (gfx1250) — compile-verified, matrix operand order checked by reading, not run on hardware
//
#include <hip/hip_runtime.h>
#include <hip/hip_bf16.h>
#include <math.h>

// ---------------- problem constants (from reference setup_inputs) ------------
#define T_TOK 4096          // B*S tokens
#define H_DIM 2048
#define E_NUM 16
#define F_DIM 768
#define TOPK  4
#define R_NUM (T_TOK * TOPK) // 16384 compacted rows

typedef __bf16 bf16_t;
typedef bf16_t v16bf __attribute__((ext_vector_type(16)));
typedef bf16_t v8bf  __attribute__((ext_vector_type(8)));
typedef float  v8f   __attribute__((ext_vector_type(8)));

// Load 16 f32 from a row with the CDNA5 16x32 bf16 A/B per-lane K pattern:
// lane half (lane>>4) selects K base; elems 0..7 -> k0+half*8+i, 8..15 -> k0+16+half*8+i
__device__ __forceinline__ v16bf load_frag_row_f32(const float* __restrict__ row,
                                                   int k0, int half) {
  const float* p = row + k0 + half * 8;
  float4 f0 = ((const float4*)p)[0];
  float4 f1 = ((const float4*)p)[1];
  float4 f2 = ((const float4*)(p + 16))[0];
  float4 f3 = ((const float4*)(p + 16))[1];
  v16bf v;
  v[0]=(bf16_t)f0.x; v[1]=(bf16_t)f0.y; v[2]=(bf16_t)f0.z; v[3]=(bf16_t)f0.w;
  v[4]=(bf16_t)f1.x; v[5]=(bf16_t)f1.y; v[6]=(bf16_t)f1.z; v[7]=(bf16_t)f1.w;
  v[8]=(bf16_t)f2.x; v[9]=(bf16_t)f2.y; v[10]=(bf16_t)f2.z; v[11]=(bf16_t)f2.w;
  v[12]=(bf16_t)f3.x; v[13]=(bf16_t)f3.y; v[14]=(bf16_t)f3.z; v[15]=(bf16_t)f3.w;
  return v;
}

// A fragment from LDS-staged tile As[64 rows][32 k] (bf16, row-major)
__device__ __forceinline__ v16bf lds_a_frag(const bf16_t* As, int mt, int lane, int half) {
  const bf16_t* p = As + (mt * 16 + (lane & 15)) * 32 + half * 8;
  v8bf lo = *(const v8bf*)p;
  v8bf hi = *(const v8bf*)(p + 16);
  v16bf v;
#pragma unroll
  for (int i = 0; i < 8; i++) { v[i] = lo[i]; v[8 + i] = hi[i]; }
  return v;
}

// ---------------- K0: zero counters --------------------------------------
__global__ void moe_zero(int* counts, int* pos) {
  int i = threadIdx.x;
  if (i < E_NUM) { counts[i] = 0; pos[i] = 0; }
}

// ---------------- K1: router (logits -> softmax -> top4 -> renorm) --------
__global__ void __launch_bounds__(128)
moe_router(const float* __restrict__ x, const float* __restrict__ Wr,
           int* __restrict__ topk_id, float* __restrict__ topk_w,
           int* __restrict__ counts) {
  const int wave = threadIdx.x >> 5;
  const int lane = threadIdx.x & 31;
  const int t = blockIdx.x * 4 + wave;
  if (t >= T_TOK) return;
  const float* xr = x + (size_t)t * H_DIM;
  float s[E_NUM];
#pragma unroll
  for (int e = 0; e < E_NUM; e++) s[e] = 0.f;
  for (int h = lane; h < H_DIM; h += 32) {
    float xv = xr[h];
#pragma unroll
    for (int e = 0; e < E_NUM; e++) s[e] += xv * Wr[e * H_DIM + h];
  }
#pragma unroll
  for (int e = 0; e < E_NUM; e++)
    for (int off = 16; off > 0; off >>= 1) s[e] += __shfl_xor(s[e], off, 32);
  if (lane == 0) {
    float mx = -1e30f;
#pragma unroll
    for (int e = 0; e < E_NUM; e++) mx = fmaxf(mx, s[e]);
    float p[E_NUM];
#pragma unroll
    for (int e = 0; e < E_NUM; e++) p[e] = __expf(s[e] - mx);
    int   id[TOPK];
    float wv[TOPK];
    float tot = 0.f;
#pragma unroll
    for (int k = 0; k < TOPK; k++) {
      int best = 0; float bv = -1.f;
#pragma unroll
      for (int e = 0; e < E_NUM; e++)
        if (p[e] > bv) { bv = p[e]; best = e; }
      id[k] = best; wv[k] = bv; p[best] = -2.f; tot += bv;
    }
    float inv = 1.f / tot; // renormalized top-k softmax (den cancels)
#pragma unroll
    for (int k = 0; k < TOPK; k++) {
      topk_id[t * TOPK + k] = id[k];
      topk_w[t * TOPK + k]  = wv[k] * inv;
      atomicAdd(&counts[id[k]], 1);
    }
  }
}

// ---------------- K2: prefix sum over experts -----------------------------
__global__ void moe_offsets(const int* __restrict__ counts, int* __restrict__ offsets) {
  if (threadIdx.x == 0) {
    int acc = 0;
    for (int e = 0; e < E_NUM; e++) { offsets[e] = acc; acc += counts[e]; }
    offsets[E_NUM] = acc;
  }
}

// ---------------- K3: compact (token,k) pairs per expert ------------------
__global__ void moe_assign(const int* __restrict__ topk_id, const float* __restrict__ topk_w,
                           const int* __restrict__ offsets, int* __restrict__ pos,
                           int* __restrict__ row_token, float* __restrict__ row_w,
                           int* __restrict__ tk_row) {
  int id = blockIdx.x * blockDim.x + threadIdx.x;
  if (id >= R_NUM) return;
  int t = id >> 2;
  int e = topk_id[id];
  int slot = atomicAdd(&pos[e], 1);
  int row = offsets[e] + slot;
  row_token[row] = t;
  row_w[row]     = topk_w[id];
  tk_row[id]     = row;
}

// ---------------- K4: gate+up GEMM + SiLU (bf16 WMMA) ---------------------
// grid: (F/64, E, zChunks), block 128 (4 waves, each owns 16 N columns)
__global__ void __launch_bounds__(128)
moe_gate_up(const float* __restrict__ x, const float* __restrict__ Wg,
            const float* __restrict__ Wu, const int* __restrict__ offsets,
            const int* __restrict__ counts, const int* __restrict__ row_token,
            bf16_t* __restrict__ act) {
  __shared__ bf16_t As[64 * 32]; // 4KB staged A tile (64 rows x 32 K)
  const int e    = blockIdx.y;
  const int wave = threadIdx.x >> 5;
  const int lane = threadIdx.x & 31;
  const int half = lane >> 4;
  const int nCol = blockIdx.x * 64 + wave * 16 + (lane & 15);
  const int rowBeg = offsets[e];
  const int cnt    = counts[e];
  const int rowEnd = rowBeg + cnt;
  const int nIter  = (cnt + 63) >> 6;
  const float* gRow = Wg + ((size_t)e * F_DIM + nCol) * H_DIM;
  const float* uRow = Wu + ((size_t)e * F_DIM + nCol) * H_DIM;
  const int sRow = threadIdx.x >> 1;        // 0..63 : staged row
  const int sSeg = (threadIdx.x & 1) * 16;  // 0/16 : K half

  for (int it = blockIdx.z; it < nIter; it += gridDim.z) {
    const int r0 = rowBeg + it * 64;
    const int gRowIdx = r0 + sRow;
    const bool sValid = gRowIdx < rowEnd;
    const float* aRow = x + (size_t)row_token[sValid ? gRowIdx : rowBeg] * H_DIM;

    v8f zero = {0.f,0.f,0.f,0.f,0.f,0.f,0.f,0.f};
    v8f hg[4], hu[4];
#pragma unroll
    for (int mt = 0; mt < 4; mt++) { hg[mt] = zero; hu[mt] = zero; }

    for (int k0 = 0; k0 < H_DIM; k0 += 32) {
      __syncthreads(); // previous chunk consumed
      {
        v8bf lo, hi;
        if (sValid) {
          const float4* q = (const float4*)(aRow + k0 + sSeg);
          float4 f0 = q[0], f1 = q[1], f2 = q[2], f3 = q[3];
          lo[0]=(bf16_t)f0.x; lo[1]=(bf16_t)f0.y; lo[2]=(bf16_t)f0.z; lo[3]=(bf16_t)f0.w;
          lo[4]=(bf16_t)f1.x; lo[5]=(bf16_t)f1.y; lo[6]=(bf16_t)f1.z; lo[7]=(bf16_t)f1.w;
          hi[0]=(bf16_t)f2.x; hi[1]=(bf16_t)f2.y; hi[2]=(bf16_t)f2.z; hi[3]=(bf16_t)f2.w;
          hi[4]=(bf16_t)f3.x; hi[5]=(bf16_t)f3.y; hi[6]=(bf16_t)f3.z; hi[7]=(bf16_t)f3.w;
        } else {
#pragma unroll
          for (int i = 0; i < 8; i++) { lo[i] = (bf16_t)0.f; hi[i] = (bf16_t)0.f; }
        }
        v8bf* dst = (v8bf*)(As + sRow * 32 + sSeg);
        dst[0] = lo; dst[1] = hi;
      }
      __syncthreads();

      v16bf bg = load_frag_row_f32(gRow, k0, half);
      v16bf bu = load_frag_row_f32(uRow, k0, half);
#pragma unroll
      for (int mt = 0; mt < 4; mt++) {
        v16bf a = lds_a_frag(As, mt, lane, half);
        hg[mt] = __builtin_amdgcn_wmma_f32_16x16x32_bf16(
            false, a, false, bg, (short)0, hg[mt], false, false);
        hu[mt] = __builtin_amdgcn_wmma_f32_16x16x32_bf16(
            false, a, false, bu, (short)0, hu[mt], false, false);
      }
    }

    // C layout: VGPR r holds M = r + half*8 ; N = lane&15 (== nCol's lane part)
#pragma unroll
    for (int mt = 0; mt < 4; mt++) {
#pragma unroll
      for (int r = 0; r < 8; r++) {
        int row = r0 + mt * 16 + half * 8 + r;
        if (row < rowEnd) {
          float g = hg[mt][r], u = hu[mt][r];
          float v = (g / (1.f + __expf(-g))) * u; // silu(g)*u
          act[(size_t)row * F_DIM + nCol] = (bf16_t)v;
        }
      }
    }
  }
}

// ---------------- K5: down GEMM, weight applied per row -------------------
// grid: (H/64, E, zChunks), block 128
__global__ void __launch_bounds__(128)
moe_down(const bf16_t* __restrict__ act, const float* __restrict__ Wd,
         const int* __restrict__ offsets, const int* __restrict__ counts,
         const float* __restrict__ row_w, float* __restrict__ out_rows) {
  __shared__ bf16_t As[64 * 32];
  const int e    = blockIdx.y;
  const int wave = threadIdx.x >> 5;
  const int lane = threadIdx.x & 31;
  const int half = lane >> 4;
  const int nCol = blockIdx.x * 64 + wave * 16 + (lane & 15); // output H column
  const int rowBeg = offsets[e];
  const int cnt    = counts[e];
  const int rowEnd = rowBeg + cnt;
  const int nIter  = (cnt + 63) >> 6;
  const float* bRow = Wd + ((size_t)e * H_DIM + nCol) * F_DIM;
  const int sRow = threadIdx.x >> 1;
  const int sSeg = (threadIdx.x & 1) * 16;

  for (int it = blockIdx.z; it < nIter; it += gridDim.z) {
    const int r0 = rowBeg + it * 64;
    const int gRowIdx = r0 + sRow;
    const bool sValid = gRowIdx < rowEnd;
    const bf16_t* aRow = act + (size_t)(sValid ? gRowIdx : rowBeg) * F_DIM;

    v8f zero = {0.f,0.f,0.f,0.f,0.f,0.f,0.f,0.f};
    v8f c[4];
#pragma unroll
    for (int mt = 0; mt < 4; mt++) c[mt] = zero;

    for (int k0 = 0; k0 < F_DIM; k0 += 32) {
      __syncthreads();
      {
        v8bf lo, hi;
        if (sValid) {
          const v8bf* q = (const v8bf*)(aRow + k0 + sSeg);
          lo = q[0]; hi = q[1];
        } else {
#pragma unroll
          for (int i = 0; i < 8; i++) { lo[i] = (bf16_t)0.f; hi[i] = (bf16_t)0.f; }
        }
        v8bf* dst = (v8bf*)(As + sRow * 32 + sSeg);
        dst[0] = lo; dst[1] = hi;
      }
      __syncthreads();

      v16bf b = load_frag_row_f32(bRow, k0, half);
#pragma unroll
      for (int mt = 0; mt < 4; mt++) {
        v16bf a = lds_a_frag(As, mt, lane, half);
        c[mt] = __builtin_amdgcn_wmma_f32_16x16x32_bf16(
            false, a, false, b, (short)0, c[mt], false, false);
      }
    }

#pragma unroll
    for (int mt = 0; mt < 4; mt++) {
#pragma unroll
      for (int r = 0; r < 8; r++) {
        int row = r0 + mt * 16 + half * 8 + r;
        if (row < rowEnd)
          out_rows[(size_t)row * H_DIM + nCol] = c[mt][r] * row_w[row];
      }
    }
  }
}

// ---------------- K6: deterministic combine out[t] = sum_k rows -----------
__global__ void moe_combine(const float* __restrict__ out_rows,
                            const int* __restrict__ tk_row,
                            float* __restrict__ out) {
  const int nV = (T_TOK * H_DIM) / 4;
  int id = blockIdx.x * blockDim.x + threadIdx.x;
  if (id >= nV) return;
  int t  = id / (H_DIM / 4);
  int c4 = id % (H_DIM / 4);
  float4 acc = {0.f, 0.f, 0.f, 0.f};
#pragma unroll
  for (int k = 0; k < TOPK; k++) {
    int row = tk_row[t * TOPK + k];
    float4 v = ((const float4*)(out_rows + (size_t)row * H_DIM))[c4];
    acc.x += v.x; acc.y += v.y; acc.z += v.z; acc.w += v.w;
  }
  ((float4*)out)[id] = acc;
}

// ---------------- host-side launch ----------------------------------------
extern "C" void kernel_launch(void* const* d_in, const int* in_sizes, int n_in,
                              void* d_out, int out_size, void* d_ws, size_t ws_size,
                              hipStream_t stream) {
  (void)in_sizes; (void)n_in; (void)out_size; (void)ws_size;
  const float* x  = (const float*)d_in[0]; // [T,H]
  const float* Wr = (const float*)d_in[1]; // [E,H]
  const float* Wg = (const float*)d_in[2]; // [E,F,H]
  const float* Wu = (const float*)d_in[3]; // [E,F,H]
  const float* Wd = (const float*)d_in[4]; // [E,H,F]
  float* out = (float*)d_out;

  char* ws = (char*)d_ws;
  size_t cur = 0;
  auto alloc = [&](size_t bytes) -> void* {
    cur = (cur + 255) & ~(size_t)255;
    void* p = ws + cur; cur += bytes; return p;
  };
  int*    topk_id   = (int*)   alloc((size_t)R_NUM * sizeof(int));
  float*  topk_w    = (float*) alloc((size_t)R_NUM * sizeof(float));
  int*    counts    = (int*)   alloc(E_NUM * sizeof(int));
  int*    offsets   = (int*)   alloc((E_NUM + 1) * sizeof(int));
  int*    pos       = (int*)   alloc(E_NUM * sizeof(int));
  int*    row_token = (int*)   alloc((size_t)R_NUM * sizeof(int));
  float*  row_w     = (float*) alloc((size_t)R_NUM * sizeof(float));
  int*    tk_row    = (int*)   alloc((size_t)R_NUM * sizeof(int));
  bf16_t* act       = (bf16_t*)alloc((size_t)R_NUM * F_DIM * sizeof(bf16_t));
  float*  out_rows  = (float*) alloc((size_t)R_NUM * H_DIM * sizeof(float));

  moe_zero<<<1, 32, 0, stream>>>(counts, pos);
  moe_router<<<T_TOK / 4, 128, 0, stream>>>(x, Wr, topk_id, topk_w, counts);
  moe_offsets<<<1, 32, 0, stream>>>(counts, offsets);
  moe_assign<<<R_NUM / 256, 256, 0, stream>>>(topk_id, topk_w, offsets, pos,
                                              row_token, row_w, tk_row);
  moe_gate_up<<<dim3(F_DIM / 64, E_NUM, 8), 128, 0, stream>>>(
      x, Wg, Wu, offsets, counts, row_token, act);
  moe_down<<<dim3(H_DIM / 64, E_NUM, 8), 128, 0, stream>>>(
      act, Wd, offsets, counts, row_w, out_rows);
  moe_combine<<<(T_TOK * H_DIM / 4 + 255) / 256, 256, 0, stream>>>(
      out_rows, tk_row, out);
}
